// EnhancedTransformerBlock_79336635892684
// MI455X (gfx1250) — matrix operand order repaired and checked by reading, not compiled, not run on hardware
//
#include <hip/hip_runtime.h>
#include <math.h>

// ---------------------------------------------------------------------------
// Types / helpers
// ---------------------------------------------------------------------------
typedef __attribute__((ext_vector_type(16))) __bf16 v16bf;
typedef __attribute__((ext_vector_type(8)))  float  v8f;
typedef unsigned short bfu;   // bf16 storage type (host-safe)

__device__ __forceinline__ bfu f2bf(float f) {
  unsigned u = __builtin_bit_cast(unsigned, f);
  u += 0x7fffu + ((u >> 16) & 1u);          // round-to-nearest-even
  return (bfu)(u >> 16);
}

union Frag16 { v16bf v; unsigned u[8]; };

// A-matrix fragment (16x32 bf16): lane = row (lane&15); per ISA table the
// K pairs in VGPR j sit at K = 2j + (j>=4 ? 8 : 0) + 8*(lane>>4).
__device__ __forceinline__ v16bf frag_ld_a(const bfu* rowptr, int half) {
  Frag16 f;
#pragma unroll
  for (int j = 0; j < 8; ++j) {
    int kb = 2 * j + ((j >= 4) ? 8 : 0) + 8 * half;
    f.u[j] = *reinterpret_cast<const unsigned*>(rowptr + kb);
  }
  return f.v;
}

// B-matrix fragment (32x16 bf16): lane = column (lane&15); same K order as A:
// K pairs in VGPR j sit at K = 2j + (j>=4 ? 8 : 0) + 8*(lane>>4).
__device__ __forceinline__ v16bf frag_ld_b(const bfu* colptr, int half) {
  Frag16 f;
#pragma unroll
  for (int j = 0; j < 8; ++j) {
    int kb = 2 * j + ((j >= 4) ? 8 : 0) + 8 * half;
    f.u[j] = *reinterpret_cast<const unsigned*>(colptr + kb);
  }
  return f.v;
}

__device__ __forceinline__ v8f wmma_bf16(v16bf a, v16bf b, v8f c) {
  return __builtin_amdgcn_wmma_f32_16x16x32_bf16(
      /*neg_a=*/false, a, /*neg_b=*/false, b,
      /*c_mod=*/(short)0, c, /*reuse_a=*/false, /*reuse_b=*/false);
}

// ds_swizzle xor-butterfly (group-of-32 mode): offset = xor<<10 | and(0x1F).
// One DS op with an immediate pattern -- no index VGPRs, no bpermute.
template <int IMM>
__device__ __forceinline__ float swz(float v) {
  int i = __builtin_amdgcn_ds_swizzle(__builtin_bit_cast(int, v), IMM);
  return __builtin_bit_cast(float, i);
}

// ---------------------------------------------------------------------------
// Weight convert + transpose: w f32 [K][Nt] row-major -> wt bf16 [Nt][K]
// ---------------------------------------------------------------------------
__global__ void k_transpose_w(const float* __restrict__ w, bfu* __restrict__ wt,
                              int K, int Nt) {
  int o = blockIdx.x * blockDim.x + threadIdx.x;
  if (o >= K * Nt) return;
  int n = o / K;
  int k = o - n * K;
  wt[o] = f2bf(w[(size_t)k * Nt + n]);
}

// ---------------------------------------------------------------------------
// LayerNorm: f32 row -> bf16 row
// ---------------------------------------------------------------------------
__global__ __launch_bounds__(256) void k_layernorm(
    const float* __restrict__ x, const float* __restrict__ g,
    const float* __restrict__ b, bfu* __restrict__ out, int C) {
  __shared__ float red[256];
  const int row = blockIdx.x, tid = threadIdx.x;
  const float* xr = x + (size_t)row * C;
  float s = 0.f, ss = 0.f;
  for (int i = tid; i < C; i += 256) { float v = xr[i]; s += v; ss += v * v; }
  red[tid] = s; __syncthreads();
  for (int w = 128; w > 0; w >>= 1) { if (tid < w) red[tid] += red[tid + w]; __syncthreads(); }
  float mu = red[0] / C;
  __syncthreads();
  red[tid] = ss; __syncthreads();
  for (int w = 128; w > 0; w >>= 1) { if (tid < w) red[tid] += red[tid + w]; __syncthreads(); }
  float var  = red[0] / C - mu * mu;
  float rstd = rsqrtf(var + 1e-5f);
  bfu* orow = out + (size_t)row * C;
  for (int i = tid; i < C; i += 256)
    orow[i] = f2bf((xr[i] - mu) * rstd * g[i] + b[i]);
}

// ---------------------------------------------------------------------------
// Tiled WMMA GEMM: D = A[M,K](bf16) * Bt[Nt,K]^T(bf16) + bias, epilogues:
//   EPI 0: bias              -> bf16 out
//   EPI 1: bias + residual   -> f32 out
//   EPI 2: bias + exact GELU -> bf16 out
// Block: 256 threads = 8 waves; tile 128(M) x 64(N); K staged 64 via LDS,
// so each barrier pair covers 8 WMMAs.
// ---------------------------------------------------------------------------
template <int EPI>
__global__ __launch_bounds__(256) void k_gemm_bf16(
    const bfu* __restrict__ A, const bfu* __restrict__ Bt,
    const float* __restrict__ bias, const float* __restrict__ resid,
    float* __restrict__ outf, bfu* __restrict__ outb,
    int M, int Nt, int K) {
  __shared__ bfu As[128][72];   // 64 K-chunk, stride 72 to spread banks
  __shared__ bfu Bs[64][72];

  const int tid  = threadIdx.x;
  const int lane = tid & 31;
  const int wave = tid >> 5;
  const int half = lane >> 4, lr = lane & 15;
  const int wm = wave & 3, wn = wave >> 2;        // 4x2 wave grid
  const int m0 = blockIdx.x * 128;
  const int n0 = blockIdx.y * 64;

  const int arow = tid >> 1, aseg = tid & 1;      // 2 threads/row, 32 elems each
  const int brow = tid >> 2, bseg = tid & 3;      // 4 threads/row, 16 elems each
  int ar = m0 + arow; if (ar > M - 1) ar = M - 1; // clamp rows for edge tile
  const bfu* aptr = A  + (size_t)ar * K + aseg * 32;
  const bfu* bptr = Bt + (size_t)(n0 + brow) * K + bseg * 16;

  const v8f zero8 = {0.f, 0.f, 0.f, 0.f, 0.f, 0.f, 0.f, 0.f};
  v8f acc[2][2] = {{zero8, zero8}, {zero8, zero8}};

  for (int k0 = 0; k0 < K; k0 += 64) {
    __syncthreads();
    {
      const uint4* s = reinterpret_cast<const uint4*>(aptr + k0);
      uint4 d0 = s[0], d1 = s[1], d2 = s[2], d3 = s[3];
      uint4* d = reinterpret_cast<uint4*>(&As[arow][aseg * 32]);
      d[0] = d0; d[1] = d1; d[2] = d2; d[3] = d3;
    }
    {
      const uint4* s = reinterpret_cast<const uint4*>(bptr + k0);
      uint4 d0 = s[0], d1 = s[1];
      uint4* d = reinterpret_cast<uint4*>(&Bs[brow][bseg * 16]);
      d[0] = d0; d[1] = d1;
    }
    if (k0 + 64 < K) {                            // pull next chunk toward L2/L0
      __builtin_prefetch(aptr + k0 + 64, 0, 0);
      __builtin_prefetch(bptr + k0 + 64, 0, 0);
    }
    __syncthreads();

#pragma unroll
    for (int ks = 0; ks < 64; ks += 32) {
      v16bf a0 = frag_ld_a(&As[wm * 32 +  0 + lr][ks], half);
      v16bf a1 = frag_ld_a(&As[wm * 32 + 16 + lr][ks], half);
      v16bf b0 = frag_ld_b(&Bs[wn * 32 +  0 + lr][ks], half);
      v16bf b1 = frag_ld_b(&Bs[wn * 32 + 16 + lr][ks], half);
      acc[0][0] = wmma_bf16(a0, b0, acc[0][0]);
      acc[0][1] = wmma_bf16(a0, b1, acc[0][1]);
      acc[1][0] = wmma_bf16(a1, b0, acc[1][0]);
      acc[1][1] = wmma_bf16(a1, b1, acc[1][1]);
    }
  }

#pragma unroll
  for (int i = 0; i < 2; ++i)
#pragma unroll
    for (int j = 0; j < 2; ++j) {
      const int col = n0 + wn * 32 + j * 16 + lr;
      const float bcol = bias[col];
#pragma unroll
      for (int r = 0; r < 8; ++r) {
        const int row = m0 + wm * 32 + i * 16 + r + 8 * half;
        if (row < M) {
          float v = acc[i][j][r] + bcol;
          size_t idx = (size_t)row * Nt + col;
          if (EPI == 0) {
            outb[idx] = f2bf(v);
          } else if (EPI == 1) {
            outf[idx] = resid[idx] + v;
          } else {
            float gel = 0.5f * v * (1.0f + erff(v * 0.70710678118654752f));
            outb[idx] = f2bf(gel);
          }
        }
      }
    }
}

// ---------------------------------------------------------------------------
// Flash attention: one wave handles a 16-query tile of one (b,h).
// qkv: bf16 [B*N, 3C] with q at h*D+d, k at C+h*D+d, v at 2C+h*D+d.
// o:   bf16 [B*N, C]
// ---------------------------------------------------------------------------
__global__ __launch_bounds__(32) void k_attention(
    const bfu* __restrict__ qkv, bfu* __restrict__ o,
    int Bsz, int N, int H, int D) {
  const int C  = H * D;          // 768
  const int C3 = 3 * C;          // 2304
  const int qt = blockIdx.x, h = blockIdx.y, bb = blockIdx.z;
  const int q0 = qt * 16;
  const int lane = threadIdx.x;
  const int half = lane >> 4, lr = lane & 15;

  __shared__ bfu Qs[16][72];     // 16 queries x 64 dims (+pad)
  __shared__ bfu Ks[32][72];     // 32 keys    x 64 dims (+pad)
  __shared__ bfu Vt[64][40];     // 64 dims    x 32 keys (transposed, +pad)
  __shared__ bfu Ps[16][32];     // probabilities tile

  // Stage Q tile (rows clamped; garbage rows never stored).
  for (int idx = lane; idx < 16 * 64; idx += 32) {
    int r = idx >> 6, d = idx & 63;
    int q = q0 + r; if (q > N - 1) q = N - 1;
    Qs[r][d] = qkv[(size_t)(bb * N + q) * C3 + h * D + d];
  }
  v16bf aq0 = frag_ld_a(&Qs[lr][0],  half);
  v16bf aq1 = frag_ld_a(&Qs[lr][32], half);

  const v8f zero8 = {0.f, 0.f, 0.f, 0.f, 0.f, 0.f, 0.f, 0.f};
  v8f Oc[4] = {zero8, zero8, zero8, zero8};
  float mrow[8], lrow[8];
#pragma unroll
  for (int r = 0; r < 8; ++r) { mrow[r] = -1e30f; lrow[r] = 0.f; }
  const float scale = 0.125f;    // HEAD_DIM^-0.5, D=64

  const int nchunk = (N + 31) / 32;
  for (int ch = 0; ch < nchunk; ++ch) {
    const int kb0 = ch * 32;
    for (int idx = lane; idx < 32 * 64; idx += 32) {
      int r = idx >> 6, d = idx & 63;
      int kk = kb0 + r; if (kk > N - 1) kk = N - 1;
      Ks[r][d] = qkv[(size_t)(bb * N + kk) * C3 + C + h * D + d];
    }
    for (int idx = lane; idx < 64 * 32; idx += 32) {
      int d = idx >> 5, kr = idx & 31;
      int kk = kb0 + kr; if (kk > N - 1) kk = N - 1;
      Vt[d][kr] = qkv[(size_t)(bb * N + kk) * C3 + 2 * C + h * D + d];
    }

    // S = Q * K^T  (two 16x16 key tiles, K-dim = 64 -> 2 WMMAs each)
    v8f s0 = zero8, s1 = zero8;
    s0 = wmma_bf16(aq0, frag_ld_b(&Ks[0  + lr][0],  half), s0);
    s0 = wmma_bf16(aq1, frag_ld_b(&Ks[0  + lr][32], half), s0);
    s1 = wmma_bf16(aq0, frag_ld_b(&Ks[16 + lr][0],  half), s1);
    s1 = wmma_bf16(aq1, frag_ld_b(&Ks[16 + lr][32], half), s1);

    const bool inv0 = (kb0 + lr)      >= N;
    const bool inv1 = (kb0 + 16 + lr) >= N;
#pragma unroll
    for (int r = 0; r < 8; ++r) {
      float e0 = inv0 ? -1e30f : s0[r] * scale;
      float e1 = inv1 ? -1e30f : s1[r] * scale;
      // row max across the 16 lanes of this half (row = r + 8*half):
      // xor 1/2/4/8 stays inside the half -> ds_swizzle immediate patterns.
      float mx = fmaxf(e0, e1);
      mx = fmaxf(mx, swz<0x041F>(mx));
      mx = fmaxf(mx, swz<0x081F>(mx));
      mx = fmaxf(mx, swz<0x101F>(mx));
      mx = fmaxf(mx, swz<0x201F>(mx));
      float mnew  = fmaxf(mrow[r], mx);
      float alpha = __expf(mrow[r] - mnew);
      float p0 = __expf(e0 - mnew);
      float p1 = __expf(e1 - mnew);
      float rs = p0 + p1;
      rs += swz<0x041F>(rs);
      rs += swz<0x081F>(rs);
      rs += swz<0x101F>(rs);
      rs += swz<0x201F>(rs);
      lrow[r] = lrow[r] * alpha + rs;
      mrow[r] = mnew;
#pragma unroll
      for (int t = 0; t < 4; ++t) Oc[t][r] *= alpha;
      Ps[r + 8 * half][lr]      = f2bf(p0);   // C-layout -> row-major LDS
      Ps[r + 8 * half][16 + lr] = f2bf(p1);
    }

    // O += P * V   (A = P[16x32 keys], B = Vt, 4 d-tiles)
    v16bf ap = frag_ld_a(&Ps[lr][0], half);
#pragma unroll
    for (int t = 0; t < 4; ++t)
      Oc[t] = wmma_bf16(ap, frag_ld_b(&Vt[t * 16 + lr][0], half), Oc[t]);
  }

#pragma unroll
  for (int r = 0; r < 8; ++r) {
    const int q = q0 + r + 8 * half;
    const float li = 1.0f / lrow[r];
#pragma unroll
    for (int t = 0; t < 4; ++t) {
      if (q < N)
        o[(size_t)(bb * N + q) * C + h * D + t * 16 + lr] =
            f2bf(Oc[t][r] * li);
    }
  }
}

// ---------------------------------------------------------------------------
// Launcher
// ---------------------------------------------------------------------------
extern "C" void kernel_launch(void* const* d_in, const int* in_sizes, int n_in,
                              void* d_out, int out_size, void* d_ws, size_t ws_size,
                              hipStream_t stream) {
  (void)in_sizes; (void)n_in; (void)out_size; (void)ws_size;
  const float* x      = (const float*)d_in[0];
  const float* ln1_g  = (const float*)d_in[1];
  const float* ln1_b  = (const float*)d_in[2];
  const float* qkv_w  = (const float*)d_in[3];
  const float* qkv_b  = (const float*)d_in[4];
  const float* proj_w = (const float*)d_in[5];
  const float* proj_b = (const float*)d_in[6];
  const float* ln2_g  = (const float*)d_in[7];
  const float* ln2_b  = (const float*)d_in[8];
  const float* fc1_w  = (const float*)d_in[9];
  const float* fc1_b  = (const float*)d_in[10];
  const float* fc2_w  = (const float*)d_in[11];
  const float* fc2_b  = (const float*)d_in[12];
  float* out = (float*)d_out;

  const int Bsz = 32, N = 577, C = 768, H = 12, D = 64;
  const int C3 = 3 * C, HID = 4 * C;
  const int M = Bsz * N;                                   // 18464 rows

  char* ws = (char*)d_ws;
  size_t off = 0;
  auto take = [&](size_t bytes) {
    char* p = ws + off;
    off += (bytes + 255) & ~(size_t)255;
    return p;
  };
  bfu* h_ws    = (bfu*)take((size_t)M * C  * 2);  // ln1 out; reused for ln2 out
  bfu* qkv_ws  = (bfu*)take((size_t)M * C3 * 2);
  bfu* o_ws    = (bfu*)take((size_t)M * C  * 2);
  bfu* act_ws  = (bfu*)take((size_t)M * HID * 2);
  bfu* wt_qkv  = (bfu*)take((size_t)C3 * C * 2);
  bfu* wt_proj = (bfu*)take((size_t)C  * C * 2);
  bfu* wt_fc1  = (bfu*)take((size_t)HID * C * 2);
  bfu* wt_fc2  = (bfu*)take((size_t)C * HID * 2);

  // Weights -> bf16, transposed to [N][K] so both GEMM operands are K-major.
  { int n = C * C3;   k_transpose_w<<<(n + 255) / 256, 256, 0, stream>>>(qkv_w,  wt_qkv,  C,   C3); }
  { int n = C * C;    k_transpose_w<<<(n + 255) / 256, 256, 0, stream>>>(proj_w, wt_proj, C,   C);  }
  { int n = C * HID;  k_transpose_w<<<(n + 255) / 256, 256, 0, stream>>>(fc1_w,  wt_fc1,  C,   HID);}
  { int n = HID * C;  k_transpose_w<<<(n + 255) / 256, 256, 0, stream>>>(fc2_w,  wt_fc2,  HID, C);  }

  // LN1
  k_layernorm<<<M, 256, 0, stream>>>(x, ln1_g, ln1_b, h_ws, C);

  // QKV = LN1(x) @ qkv_w + qkv_b        (bf16 out)
  dim3 gq((M + 127) / 128, C3 / 64);
  k_gemm_bf16<0><<<gq, 256, 0, stream>>>(h_ws, wt_qkv, qkv_b, nullptr,
                                         nullptr, qkv_ws, M, C3, C);

  // Flash attention
  dim3 ga((N + 15) / 16, H, Bsz);
  k_attention<<<ga, 32, 0, stream>>>(qkv_ws, o_ws, Bsz, N, H, D);

  // x1 = x + o @ proj_w + proj_b        (f32 out -> d_out)
  dim3 gp((M + 127) / 128, C / 64);
  k_gemm_bf16<1><<<gp, 256, 0, stream>>>(o_ws, wt_proj, proj_b, x,
                                         out, nullptr, M, C, C);

  // LN2
  k_layernorm<<<M, 256, 0, stream>>>(out, ln2_g, ln2_b, h_ws, C);

  // act = GELU(LN2(x1) @ fc1_w + fc1_b) (bf16 out)
  dim3 g1((M + 127) / 128, HID / 64);
  k_gemm_bf16<2><<<g1, 256, 0, stream>>>(h_ws, wt_fc1, fc1_b, nullptr,
                                         nullptr, act_ws, M, HID, C);

  // out = x1 + act @ fc2_w + fc2_b      (f32 out -> d_out, residual from d_out)
  k_gemm_bf16<1><<<gp, 256, 0, stream>>>(act_ws, wt_fc2, fc2_b, out,
                                         out, nullptr, M, C, HID);
}
